// Q_s2v_13597866459921
// MI455X (gfx1250) — matrix operand order repaired and checked by reading, not compiled, not run on hardware
//
#include <hip/hip_runtime.h>
#include <hip/hip_bf16.h>

#define NNODES 100000
#define NEDGES 1600000
#define PDIM   64
#define TSTEPS 4
#define GBATCH 64
#define NTILES (NNODES / 16)          // 6250, exact
#define CDIV(a, b) (((a) + (b) - 1) / (b))

typedef __attribute__((ext_vector_type(16))) __bf16 v16bf;
typedef __attribute__((ext_vector_type(8)))  float  v8f;

union ABu { v16bf v; unsigned int u[8]; };

// native f32 -> bf16 (RNE) so the backend emits v_cvt_*bf16_f32
__device__ __forceinline__ unsigned short f2bf(float f) {
  __bf16 h = (__bf16)f;
  return __builtin_bit_cast(unsigned short, h);
}

__device__ __forceinline__ void atomic_add_f32(float* p, float v) {
  __hip_atomic_fetch_add(p, v, __ATOMIC_RELAXED, __HIP_MEMORY_SCOPE_AGENT);
}

// A-matrix load from LDS (dword view). Contiguous K order matching gB:
// lanes 0-15: v0-7 = K kc+0..15 ; lanes 16-31: K kc+16..31.
__device__ __forceinline__ v16bf ldsA(const unsigned int* ls, int strideDw,
                                      int kcBase, int lane) {
  const int m  = lane & 15;
  const int hi = lane >> 4;
  const unsigned int* row = ls + m * strideDw + (kcBase >> 1) + hi * 8;
  ABu a;
#pragma unroll
  for (int i = 0; i < 4; ++i) a.u[i] = row[i];
#pragma unroll
  for (int i = 0; i < 4; ++i) a.u[4 + i] = row[4 + i];
  return a.v;
}

// B-matrix load from global bf16 weights stored row-major [rows][K] (Y = X@W^T).
// Contiguous K order (same as ldsA), not AMD's table order: lanes 0-15 hold K kc+0..15, lanes 16-31 hold K kc+16..31.
__device__ __forceinline__ v16bf gB(const unsigned int* W, int Kdw,
                                    int ntBase, int kcBase, int lane) {
  const int n  = ntBase + (lane & 15);
  const int hi = lane >> 4;
  const unsigned int* row = W + n * Kdw + (kcBase >> 1) + hi * 8;
  ABu b;
#pragma unroll
  for (int i = 0; i < 8; ++i) b.u[i] = row[i];
  return b.v;
}

__device__ __forceinline__ v8f wmma_bf16(v16bf a, v16bf b, v8f c) {
  return __builtin_amdgcn_wmma_f32_16x16x32_bf16(false, a, false, b,
                                                 (short)0, c, false, false);
}

// ---------------------------------------------------------------- utilities
__global__ __launch_bounds__(256) void s2v_zero(float* p, long long n) {
  long long i = (long long)blockIdx.x * blockDim.x + threadIdx.x;
  if (i < n) p[i] = 0.f;
}

__global__ __launch_bounds__(256) void s2v_cvt_bf16(const float* __restrict__ s,
                                                    unsigned short* __restrict__ d,
                                                    int n) {
  int i = blockIdx.x * blockDim.x + threadIdx.x;
  if (i < n) d[i] = f2bf(s[i]);
}

// c3[t][p] = sum_k relu(W4[t][k]) * W3[t][p][k]   (exact since weight >= 0)
__global__ __launch_bounds__(256) void s2v_c3(const float* __restrict__ W3,
                                              const float* __restrict__ W4,
                                              float* __restrict__ c3) {
  int i = blockIdx.x * blockDim.x + threadIdx.x;
  if (i >= TSTEPS * PDIM) return;
  int t = i >> 6, p = i & 63;
  float acc = 0.f;
  for (int k = 0; k < PDIM; ++k)
    acc += fmaxf(W4[t * PDIM + k], 0.f) * W3[((size_t)t * PDIM + p) * PDIM + k];
  c3[i] = acc;
}

// per-node incident weight sums (both directions), once per launch
__global__ __launch_bounds__(256) void s2v_sw(const long long* __restrict__ ei,
                                              const float* __restrict__ w,
                                              float* __restrict__ sw_in,
                                              float* __restrict__ sw_out) {
  int e = blockIdx.x * blockDim.x + threadIdx.x;
  if (e >= NEDGES) return;
  float v = w[e];
  atomic_add_f32(&sw_in[(size_t)ei[NEDGES + e]], v);   // dst = ei1
  atomic_add_f32(&sw_out[(size_t)ei[e]], v);           // dst = ei0
}

// edge gather + scatter-add of mu, both directions.
// 16 edges per 256-block; endpoints staged via LDS (loaded once, not 64x);
// each thread moves 4 columns with b128 loads; atomics are the required traffic.
__global__ __launch_bounds__(256) void s2v_edge(const long long* __restrict__ ei,
                                                const float* __restrict__ mu,
                                                float* __restrict__ agg_in,
                                                float* __restrict__ agg_out) {
  __shared__ int ls_s[16], ls_d[16];
  const int tid = threadIdx.x;
  const long long ebase = (long long)blockIdx.x * 16;
  if (tid < 16) ls_s[tid] = (int)ei[ebase + tid];                  // ei0
  else if (tid < 32) ls_d[tid - 16] = (int)ei[NEDGES + ebase + (tid - 16)]; // ei1
  __syncthreads();
  const int le = tid >> 4;            // edge slot 0..15
  const int q  = (tid & 15) * 4;      // column base 0..60
  const size_t s = (size_t)ls_s[le];
  const size_t d = (size_t)ls_d[le];
  const float4 vs = *(const float4*)(mu + s * 64 + q);   // mu[src] row chunk
  const float4 vd = *(const float4*)(mu + d * 64 + q);   // mu[dst] row chunk
  float* ai = agg_in  + d * 64 + q;
  float* ao = agg_out + s * 64 + q;
  atomic_add_f32(ai + 0, vs.x); atomic_add_f32(ai + 1, vs.y);
  atomic_add_f32(ai + 2, vs.z); atomic_add_f32(ai + 3, vs.w);
  atomic_add_f32(ao + 0, vd.x); atomic_add_f32(ao + 1, vd.y);
  atomic_add_f32(ao + 2, vd.z); atomic_add_f32(ao + 3, vd.w);
}

// ---------------------------------------------------------------- node tiles
#define HSd 100   // h row stride in dwords (192 bf16 cols + pad)
#define GSd 36    // g / agg row stride in dwords (64 cols + pad)
#define CSd 68    // cat row stride in dwords (128 cols + pad)

__global__ __launch_bounds__(32) void s2v_node(
    const float* __restrict__ x, const float* __restrict__ sw_in,
    const float* __restrict__ sw_out,
    const float* __restrict__ agg_in, const float* __restrict__ agg_out,
    const float* __restrict__ W1t, const float* __restrict__ c3t,
    const unsigned int* __restrict__ W2bf, const unsigned int* __restrict__ Wc1bf,
    const unsigned int* __restrict__ Wc2bf, const unsigned int* __restrict__ Wcompbf,
    float* __restrict__ mu) {
  __shared__ __align__(16) unsigned short ls_h[16 * 2 * HSd];
  __shared__ __align__(16) unsigned short ls_g[16 * 2 * GSd];
  __shared__ __align__(16) unsigned short ls_cat[16 * 2 * CSd];
  __shared__ float ls_x[16], ls_sw[2][16], ls_w1[64], ls_c3[64];

  const int lane = threadIdx.x;
  const int base = blockIdx.x * 16;
  const int lo = lane & 15, hi = lane >> 4;

  if (lane < 16) {
    ls_x[lane]     = x[base + lane];
    ls_sw[0][lane] = sw_in[base + lane];
    ls_sw[1][lane] = sw_out[base + lane];
  }
  ls_w1[lane] = W1t[lane];  ls_w1[lane + 32] = W1t[lane + 32];
  ls_c3[lane] = c3t[lane];  ls_c3[lane + 32] = c3t[lane + 32];
  __syncthreads();

  const unsigned int* h_dw = (const unsigned int*)ls_h;
  const unsigned int* g_dw = (const unsigned int*)ls_g;
  const unsigned int* c_dw = (const unsigned int*)ls_cat;

  for (int dir = 0; dir < 2; ++dir) {
    const float* agg = dir ? agg_out : agg_in;
    // stage agg tile (bf16) + rank-1 part1/part3 columns of h
    for (int i = lane; i < 16 * 64; i += 32) {
      int m = i >> 6, p = i & 63;
      ls_g[m * (2 * GSd) + p]       = f2bf(agg[(size_t)(base + m) * 64 + p]);
      ls_h[m * (2 * HSd) + p]       = f2bf(ls_x[m] * ls_w1[p]);        // part1
      ls_h[m * (2 * HSd) + 128 + p] = f2bf(ls_sw[dir][m] * ls_c3[p]);  // part3
    }
    __syncthreads();
    // part2 = agg @ W2^T  -> h columns 64..127
    for (int nt = 0; nt < 4; ++nt) {
      v8f acc = {};
#pragma unroll
      for (int kc = 0; kc < 2; ++kc)
        acc = wmma_bf16(ldsA(g_dw, GSd, kc * 32, lane),
                        gB(W2bf, 32, nt * 16, kc * 32, lane), acc);
#pragma unroll
      for (int r = 0; r < 8; ++r)
        ls_h[(r + hi * 8) * (2 * HSd) + 64 + nt * 16 + lo] = f2bf(acc[r]);
    }
    __syncthreads();
    // g = relu(h @ Wc1^T)   [16x192] x [192->64]
    for (int nt = 0; nt < 4; ++nt) {
      v8f acc = {};
#pragma unroll
      for (int kc = 0; kc < 6; ++kc)
        acc = wmma_bf16(ldsA(h_dw, HSd, kc * 32, lane),
                        gB(Wc1bf, 96, nt * 16, kc * 32, lane), acc);
#pragma unroll
      for (int r = 0; r < 8; ++r)
        ls_g[(r + hi * 8) * (2 * GSd) + nt * 16 + lo] = f2bf(fmaxf(acc[r], 0.f));
    }
    __syncthreads();
    // mu_dir = relu(g @ Wc2^T) -> cat columns [dir*64, dir*64+64)
    for (int nt = 0; nt < 4; ++nt) {
      v8f acc = {};
#pragma unroll
      for (int kc = 0; kc < 2; ++kc)
        acc = wmma_bf16(ldsA(g_dw, GSd, kc * 32, lane),
                        gB(Wc2bf, 32, nt * 16, kc * 32, lane), acc);
#pragma unroll
      for (int r = 0; r < 8; ++r)
        ls_cat[(r + hi * 8) * (2 * CSd) + dir * 64 + nt * 16 + lo] =
            f2bf(fmaxf(acc[r], 0.f));
    }
    __syncthreads();
  }
  // mu = relu(cat @ Wcompress^T)   [16x128] x [128->64]
  for (int nt = 0; nt < 4; ++nt) {
    v8f acc = {};
#pragma unroll
    for (int kc = 0; kc < 4; ++kc)
      acc = wmma_bf16(ldsA(c_dw, CSd, kc * 32, lane),
                      gB(Wcompbf, 64, nt * 16, kc * 32, lane), acc);
#pragma unroll
    for (int r = 0; r < 8; ++r)
      mu[(size_t)(base + r + hi * 8) * 64 + nt * 16 + lo] = fmaxf(acc[r], 0.f);
  }
}

// ---------------------------------------------------------------- readout
__global__ __launch_bounds__(256) void s2v_pool(const float* __restrict__ mu,
                                                const long long* __restrict__ bid,
                                                float* __restrict__ pool_sum) {
  int i = blockIdx.x * blockDim.x + threadIdx.x;
  if (i >= NNODES * 64) return;
  int n = i >> 6, p = i & 63;
  atomic_add_f32(&pool_sum[(size_t)bid[n] * 64 + p], mu[i]);
}

__global__ __launch_bounds__(256) void s2v_poolw6(const float* __restrict__ ps,
                                                  const float* __restrict__ W6,
                                                  float* __restrict__ pw) {
  int i = blockIdx.x * blockDim.x + threadIdx.x;
  if (i >= GBATCH * 64) return;
  int g = i >> 6, p = i & 63;
  float a = 0.f;
  for (int k = 0; k < 64; ++k) a += ps[g * 64 + k] * W6[p * 64 + k];
  pw[i] = a;
}

__global__ __launch_bounds__(256) void s2v_final(const float* __restrict__ mu,
                                                 const long long* __restrict__ bid,
                                                 const float* __restrict__ pw,
                                                 const float* __restrict__ W7,
                                                 const float* __restrict__ W5,
                                                 float* __restrict__ out) {
  int n = blockIdx.x * blockDim.x + threadIdx.x;
  if (n >= NNODES) return;
  const float* m = mu + (size_t)n * 64;
  size_t b = (size_t)bid[n];
  float acc = 0.f;
  for (int j = 0; j < 64; ++j) acc += fmaxf(pw[b * 64 + j], 0.f) * W5[j];
  for (int j = 0; j < 64; ++j) {
    float d = 0.f;
    for (int k = 0; k < 64; ++k) d += m[k] * W7[j * 64 + k];
    acc += fmaxf(d, 0.f) * W5[64 + j];
  }
  out[n] = fmaxf(acc, 0.f);
}

// ---------------------------------------------------------------- launcher
extern "C" void kernel_launch(void* const* d_in, const int* in_sizes, int n_in,
                              void* d_out, int out_size, void* d_ws, size_t ws_size,
                              hipStream_t stream) {
  (void)in_sizes; (void)n_in; (void)out_size; (void)ws_size;
  const float*     x      = (const float*)d_in[0];
  const float*     weight = (const float*)d_in[1];
  const long long* ei     = (const long long*)d_in[2];
  const long long* bid    = (const long long*)d_in[3];
  const float*     W1     = (const float*)d_in[4];
  const float*     W2     = (const float*)d_in[5];
  const float*     W3     = (const float*)d_in[6];
  const float*     W4     = (const float*)d_in[7];
  const float*     Wc1    = (const float*)d_in[8];
  const float*     Wc2    = (const float*)d_in[9];
  const float*     Wcomp  = (const float*)d_in[10];
  const float*     W5     = (const float*)d_in[11];
  const float*     W6     = (const float*)d_in[12];
  const float*     W7     = (const float*)d_in[13];
  float* out = (float*)d_out;

  char* ws = (char*)d_ws;
  float* mu       = (float*)ws; ws += sizeof(float) * (size_t)NNODES * PDIM;
  float* agg_in   = (float*)ws; ws += sizeof(float) * (size_t)NNODES * PDIM;
  float* agg_out  = (float*)ws; ws += sizeof(float) * (size_t)NNODES * PDIM;
  float* sw_in    = (float*)ws; ws += sizeof(float) * NNODES;
  float* sw_out   = (float*)ws; ws += sizeof(float) * NNODES;
  float* c3       = (float*)ws; ws += sizeof(float) * TSTEPS * PDIM;
  float* pool_sum = (float*)ws; ws += sizeof(float) * GBATCH * PDIM;
  float* poolw6   = (float*)ws; ws += sizeof(float) * GBATCH * PDIM;
  unsigned short* W2bf   = (unsigned short*)ws; ws += 2ull * TSTEPS * PDIM * PDIM;
  unsigned short* Wc1bf  = (unsigned short*)ws; ws += 2ull * TSTEPS * PDIM * 3 * PDIM;
  unsigned short* Wc2bf  = (unsigned short*)ws; ws += 2ull * TSTEPS * PDIM * PDIM;
  unsigned short* Wcompbf = (unsigned short*)ws;

  const long long NP = (long long)NNODES * PDIM;

  // zero-init state (zero kernels: graph-capture safe)
  s2v_zero<<<CDIV(NP, 256), 256, 0, stream>>>(mu, NP);
  s2v_zero<<<CDIV(NNODES, 256), 256, 0, stream>>>(sw_in, NNODES);
  s2v_zero<<<CDIV(NNODES, 256), 256, 0, stream>>>(sw_out, NNODES);
  s2v_zero<<<CDIV(GBATCH * PDIM, 256), 256, 0, stream>>>(pool_sum, GBATCH * PDIM);

  // bf16 weight conversion + c3 + sw
  s2v_cvt_bf16<<<CDIV(TSTEPS * PDIM * PDIM, 256), 256, 0, stream>>>(W2, W2bf, TSTEPS * PDIM * PDIM);
  s2v_cvt_bf16<<<CDIV(TSTEPS * PDIM * 3 * PDIM, 256), 256, 0, stream>>>(Wc1, Wc1bf, TSTEPS * PDIM * 3 * PDIM);
  s2v_cvt_bf16<<<CDIV(TSTEPS * PDIM * PDIM, 256), 256, 0, stream>>>(Wc2, Wc2bf, TSTEPS * PDIM * PDIM);
  s2v_cvt_bf16<<<CDIV(PDIM * 2 * PDIM, 256), 256, 0, stream>>>(Wcomp, Wcompbf, PDIM * 2 * PDIM);
  s2v_c3<<<1, 256, 0, stream>>>(W3, W4, c3);
  s2v_sw<<<CDIV(NEDGES, 256), 256, 0, stream>>>(ei, weight, sw_in, sw_out);

  for (int t = 0; t < TSTEPS; ++t) {
    s2v_zero<<<CDIV(NP, 256), 256, 0, stream>>>(agg_in, NP);
    s2v_zero<<<CDIV(NP, 256), 256, 0, stream>>>(agg_out, NP);
    s2v_edge<<<NEDGES / 16, 256, 0, stream>>>(ei, mu, agg_in, agg_out);
    s2v_node<<<NTILES, 32, 0, stream>>>(
        x, sw_in, sw_out, agg_in, agg_out,
        W1 + t * PDIM, c3 + t * PDIM,
        (const unsigned int*)(W2bf + (size_t)t * PDIM * PDIM),
        (const unsigned int*)(Wc1bf + (size_t)t * PDIM * 3 * PDIM),
        (const unsigned int*)(Wc2bf + (size_t)t * PDIM * PDIM),
        (const unsigned int*)Wcompbf, mu);
  }

  s2v_pool<<<CDIV(NNODES * 64, 256), 256, 0, stream>>>(mu, bid, pool_sum);
  s2v_poolw6<<<CDIV(GBATCH * 64, 256), 256, 0, stream>>>(pool_sum, W6, poolw6);
  s2v_final<<<CDIV(NNODES, 256), 256, 0, stream>>>(mu, bid, poolw6, W7, W5, out);
}
